// Attention_31069793419823
// MI455X (gfx1250) — matrix operand order repaired and checked by reading, not compiled, not run on hardware
//
#include <hip/hip_runtime.h>
#include <hip/hip_bf16.h>

#define BB 4
#define SS 2048
#define DM 512
#define HH 8
#define DH 64
#define NEG_BIG 1e10f

typedef __attribute__((ext_vector_type(16))) _Float16 v16h;
typedef __attribute__((ext_vector_type(2)))  _Float16 v2h;
typedef __attribute__((ext_vector_type(8)))  float    v8f;
typedef __attribute__((ext_vector_type(4)))  int      v4i;

__device__ __forceinline__ v8f wmma16(v16h a, v16h b, v8f c) {
  // D = A(16x32 f16) * B(32x16 f16) + C(16x16 f32)
  return __builtin_amdgcn_wmma_f32_16x16x32_f16(false, a, false, b,
                                                (short)0, c, false, false);
}

// ---- CDNA5 async global->LDS copy (ASYNCcnt path), with safe fallback ------
__device__ __forceinline__ void async_cp16(_Float16* l, const _Float16* g) {
#if __has_builtin(__builtin_amdgcn_global_load_async_to_lds_b128)
  typedef __attribute__((address_space(1))) v4i g_v4i;   // global <4 x i32>
  typedef __attribute__((address_space(3))) v4i l_v4i;   // LDS    <4 x i32>
  __builtin_amdgcn_global_load_async_to_lds_b128(
      (g_v4i*)g, (l_v4i*)l, 0, 0);
#else
  *(v4i*)l = *(const v4i*)g;
#endif
}
__device__ __forceinline__ void wait_async0() {
#if __has_builtin(__builtin_amdgcn_s_wait_asynccnt)
  __builtin_amdgcn_s_wait_asynccnt(0);
#else
  asm volatile("s_wait_asynccnt 0x0" ::: "memory");
#endif
}
__device__ __forceinline__ void wait_async4() {
#if __has_builtin(__builtin_amdgcn_s_wait_asynccnt)
  __builtin_amdgcn_s_wait_asynccnt(4);
#else
  asm volatile("s_wait_asynccnt 0x4" ::: "memory");
#endif
}

// ---------------------------------------------------------------------------
// Kernel 1: fused Q/K/V projection, fp32 -> f16 pack.
//   X[8192,512] @ W[512,512] ; block = 128(M) x 64(N=one head), 8 waves.
//   W tile converted ONCE to f16 in LDS (transposed) -> B frags are pure
//   ds_load_b128.  Q,K out [bh][S][64]; V out transposed [bh][64][S].
// ---------------------------------------------------------------------------
__global__ __launch_bounds__(256)
void proj_kernel(const float* __restrict__ q, const float* __restrict__ k,
                 const float* __restrict__ v, const float* __restrict__ Wq,
                 const float* __restrict__ Wk, const float* __restrict__ Wv,
                 _Float16* __restrict__ Qh, _Float16* __restrict__ Kh,
                 _Float16* __restrict__ VhT)
{
  __shared__ _Float16 lwh[64][40];        // W chunk transposed f16: [n][k], pad
  const int z = blockIdx.z;
  const float* X = (z == 0) ? q  : (z == 1) ? k  : v;
  const float* W = (z == 0) ? Wq : (z == 1) ? Wk : Wv;
  const int h     = blockIdx.y;
  const int mbase = blockIdx.x * 128;
  const int tid   = threadIdx.x;
  const int wv    = tid >> 5;
  const int lane  = tid & 31;
  const int g     = lane >> 4;            // 16-lane group
  const int c     = lane & 15;
  const int mrow0 = mbase + wv * 16;      // this wave's 16 output rows

  v8f acc[4] = {};                        // 4 x (16x16) tiles spanning 64 cols

  for (int kc = 0; kc < DM; kc += 32) {
    // cooperative load W[kc..kc+32)[h*64..+64), convert once, store f16 pairs
    #pragma unroll
    for (int r = 0; r < 4; ++r) {
      int e  = tid + r * 256;             // pair index 0..1023
      int nn = e & 63;
      int kp = e >> 6;                    // 0..15 (pair of K rows)
      float w0 = W[(size_t)(kc + 2 * kp) * DM + h * 64 + nn];
      float w1 = W[(size_t)(kc + 2 * kp + 1) * DM + h * 64 + nn];
      v2h p; p[0] = (_Float16)w0; p[1] = (_Float16)w1;
      *(v2h*)&lwh[nn][2 * kp] = p;
    }
    __syncthreads();

    // A fragment: lane row M=c, 16 contiguous f16 along K (k = 16g + j)
    const float* xrow = X + (size_t)(mrow0 + c) * DM + kc;
    v16h a;
    #pragma unroll
    for (int j = 0; j < 8; ++j) {
      a[j]     = (_Float16)xrow[16 * g + j];
      a[j + 8] = (_Float16)xrow[16 * g + 8 + j];
    }
    #pragma unroll
    for (int nt = 0; nt < 4; ++nt) {
      const _Float16* wc = &lwh[nt * 16 + c][16 * g]; // 16 contiguous f16
      v16h bf;
      #pragma unroll
      for (int j = 0; j < 16; ++j) bf[j] = wc[j];
      acc[nt] = wmma16(a, bf, acc[nt]);
    }
    __syncthreads();
  }

  // D layout: VGPR i -> row (i + 8g), col = nt*16 + c
  if (z < 2) {
    _Float16* O = (z == 0) ? Qh : Kh;
    #pragma unroll
    for (int nt = 0; nt < 4; ++nt)
      #pragma unroll
      for (int i = 0; i < 8; ++i) {
        int mr = mrow0 + i + 8 * g;
        int b  = mr >> 11, s = mr & (SS - 1);
        O[((size_t)(b * HH + h) * SS + s) * DH + nt * 16 + c] =
            (_Float16)acc[nt][i];
      }
  } else {
    #pragma unroll
    for (int nt = 0; nt < 4; ++nt)
      #pragma unroll
      for (int i = 0; i < 8; ++i) {
        int mr = mrow0 + i + 8 * g;
        int b  = mr >> 11, s = mr & (SS - 1);
        VhT[((size_t)(b * HH + h) * DH + nt * 16 + c) * SS + s] =
            (_Float16)acc[nt][i];
      }
  }
}

// ---------------------------------------------------------------------------
// Kernel 2: causal flash attention. 256 threads = 8 waves; wave owns 16 query
// rows.  K/V 64-key tiles staged in LDS via async-load-to-LDS, double
// buffered; all 8 waves share the staged tiles (was 8x duplicated loads).
// ---------------------------------------------------------------------------
__global__ __launch_bounds__(256)
void attn_kernel(const _Float16* __restrict__ Qh, const _Float16* __restrict__ Kh,
                 const _Float16* __restrict__ VhT,
                 const float* __restrict__ v_mask, const float* __restrict__ q_mask,
                 float* __restrict__ out)
{
  __shared__ _Float16 Kbuf[2][64][72];    // [key][d], padded (144B row, 16B ok)
  __shared__ _Float16 Vbuf[2][64][72];    // [dv][key]
  __shared__ _Float16 Plds[8][16][72];    // per-wave P staging (D->A transpose)

  const int bh   = blockIdx.y;
  const int b    = bh >> 3, h = bh & 7;
  const int qb   = blockIdx.x;
  const int tid  = threadIdx.x;
  const int wv   = tid >> 5, lane = tid & 31;
  const int g    = lane >> 4, c = lane & 15;
  const int q0   = qb * 128 + wv * 16;

  const _Float16* Qp = Qh  + (size_t)bh * SS * DH;
  const _Float16* Kp = Kh  + (size_t)bh * SS * DH;
  const _Float16* Vp = VhT + (size_t)bh * DH * SS;

  // stage one 64-key K tile + V tile into LDS buffer (4 async b128 / thread)
  auto stage = [&](int buf, int kt) {
    #pragma unroll
    for (int j = 0; j < 2; ++j) {
      int cid = tid + j * 256;            // 0..511 16-byte chunks
      int row = cid >> 3;                 // 0..63
      int col = (cid & 7) * 8;            // halves
      async_cp16(&Kbuf[buf][row][col], Kp + (size_t)(kt + row) * DH + col);
      async_cp16(&Vbuf[buf][row][col], Vp + (size_t)row * SS + kt + col);
    }
  };

  // Q A-fragments, loaded once: rows q0+c, head-dim split 0..31 / 32..63
  v16h a0, a1;
  {
    const _Float16* qr = Qp + (size_t)(q0 + c) * DH;
    #pragma unroll
    for (int j = 0; j < 8; ++j) {
      a0[j]     = qr[16 * g + j];
      a0[j + 8] = qr[16 * g + 8 + j];
      a1[j]     = qr[32 + 16 * g + j];
      a1[j + 8] = qr[40 + 16 * g + j];
    }
  }

  v8f o[4] = {};
  float m[8], l[8];
  #pragma unroll
  for (int i = 0; i < 8; ++i) { m[i] = -1e30f; l[i] = 0.f; }

  const int ntiles = (qb * 128 + 128) >> 6;    // causal: keys [0, qb*128+128)
  stage(0, 0);

  for (int t = 0; t < ntiles; ++t) {
    const int kt  = t * 64;
    const int buf = t & 1;
    if (t + 1 < ntiles) { stage(buf ^ 1, kt + 64); wait_async4(); }
    else                { wait_async0(); }
    __syncthreads();                       // tile t visible to all waves

    // ----- scores: S = Q(16x64) @ K^T(64x64), 8 WMMAs, K from LDS -----
    float sc[4][8];
    v8f zero = {};
    #pragma unroll
    for (int nt = 0; nt < 4; ++nt) {
      int key = kt + nt * 16 + c;
      const _Float16* kr = &Kbuf[buf][nt * 16 + c][0];
      v16h b0, b1;
      #pragma unroll
      for (int j = 0; j < 16; ++j) {
        b0[j] = kr[16 * g + j];
        b1[j] = kr[32 + 16 * g + j];
      }
      v8f s = wmma16(a0, b0, zero);
      s     = wmma16(a1, b1, s);
      float madd = -(1.0f - v_mask[b * SS + key]) * NEG_BIG;
      #pragma unroll
      for (int i = 0; i < 8; ++i) {
        int qrow = q0 + i + 8 * g;
        float sv = s[i] * 0.125f + madd;             // 1/sqrt(64)
        sc[nt][i] = (key > qrow) ? -NEG_BIG : sv;    // causal
      }
    }

    // ----- online softmax (row lives in 16 lanes of one half-wave) -----
    #pragma unroll
    for (int i = 0; i < 8; ++i) {
      float rm = fmaxf(fmaxf(sc[0][i], sc[1][i]), fmaxf(sc[2][i], sc[3][i]));
      rm = fmaxf(rm, __shfl_xor(rm, 1));
      rm = fmaxf(rm, __shfl_xor(rm, 2));
      rm = fmaxf(rm, __shfl_xor(rm, 4));
      rm = fmaxf(rm, __shfl_xor(rm, 8));
      float mn    = fmaxf(m[i], rm);
      float alpha = __expf(m[i] - mn);
      m[i] = mn;
      l[i] *= alpha;
      #pragma unroll
      for (int nt = 0; nt < 4; ++nt) o[nt][i] *= alpha;
      float rs = 0.f;
      #pragma unroll
      for (int nt = 0; nt < 4; ++nt) {
        float p = __expf(sc[nt][i] - mn);
        sc[nt][i] = p;
        rs += p;
      }
      rs += __shfl_xor(rs, 1); rs += __shfl_xor(rs, 2);
      rs += __shfl_xor(rs, 4); rs += __shfl_xor(rs, 8);
      l[i] += rs;
    }

    // ----- D->A transpose of P via per-wave LDS -----
    #pragma unroll
    for (int nt = 0; nt < 4; ++nt)
      #pragma unroll
      for (int i = 0; i < 8; ++i)
        Plds[wv][i + 8 * g][nt * 16 + c] = (_Float16)sc[nt][i];
    __syncthreads();

    v16h pa0, pa1;
    {
      const _Float16* pr = &Plds[wv][c][0];
      #pragma unroll
      for (int j = 0; j < 8; ++j) {
        pa0[j]     = pr[16 * g + j];
        pa0[j + 8] = pr[16 * g + 8 + j];
        pa1[j]     = pr[32 + 16 * g + j];
        pa1[j + 8] = pr[40 + 16 * g + j];
      }
    }

    // ----- O += P(16x64) @ V(64x64), 8 WMMAs, V from LDS -----
    #pragma unroll
    for (int nt = 0; nt < 4; ++nt) {
      const _Float16* vr = &Vbuf[buf][nt * 16 + c][0];   // one dv row
      v16h bv0, bv1;
      #pragma unroll
      for (int j = 0; j < 16; ++j) {
        bv0[j] = vr[16 * g + j];
        bv1[j] = vr[32 + 16 * g + j];
      }
      o[nt] = wmma16(pa0, bv0, o[nt]);
      o[nt] = wmma16(pa1, bv1, o[nt]);
    }
    __syncthreads();                     // protect buf before overwrite at t+1
  }

  // ----- epilogue: 1/l, query mask, coalesced f32 stores -----
  #pragma unroll
  for (int i = 0; i < 8; ++i) {
    int qrow  = q0 + i + 8 * g;
    float inv = (l[i] > 0.f) ? (1.0f / l[i]) : 0.f;
    float f   = inv * q_mask[b * SS + qrow];
    #pragma unroll
    for (int nt = 0; nt < 4; ++nt)
      out[((size_t)b * SS + qrow) * (HH * DH) + h * DH + nt * 16 + c] =
          o[nt][i] * f;
  }
}

// ---------------------------------------------------------------------------
extern "C" void kernel_launch(void* const* d_in, const int* in_sizes, int n_in,
                              void* d_out, int out_size, void* d_ws, size_t ws_size,
                              hipStream_t stream) {
  const float* q      = (const float*)d_in[0];
  const float* k      = (const float*)d_in[1];
  const float* v      = (const float*)d_in[2];
  const float* v_mask = (const float*)d_in[3];
  const float* q_mask = (const float*)d_in[4];
  const float* Wq     = (const float*)d_in[5];
  const float* Wk     = (const float*)d_in[6];
  const float* Wv     = (const float*)d_in[7];

  const size_t per = (size_t)BB * HH * SS * DH;   // 4M halves = 8 MiB each
  _Float16* Qh  = (_Float16*)d_ws;
  _Float16* Kh  = Qh + per;
  _Float16* VhT = Kh + per;

  dim3 g1((BB * SS) / 128, HH, 3);
  proj_kernel<<<g1, 256, 0, stream>>>(q, k, v, Wq, Wk, Wv, Qh, Kh, VhT);

  dim3 g2(SS / 128, BB * HH);
  attn_kernel<<<g2, 256, 0, stream>>>(Qh, Kh, VhT, v_mask, q_mask, (float*)d_out);
}
